// GATLayer_35364760715428
// MI455X (gfx1250) — matrix operand order repaired and checked by reading, not compiled, not run on hardware
//
#include <hip/hip_runtime.h>
#include <hip/hip_bf16.h>

// ---------------------------------------------------------------------------
// GAT layer, B=4 N=2048 DIN=DOUT=256 H=4 HD=64, leaky slope 0.2
//   k_cvt_x    : x fp32 -> xb bf16                              [B*N, 256]
//   k_pack_adj : adj fp32 -> 1-bit mask via __ballot            [B,N,N/32] u32
//   k_pack_w   : W fp32 -> wpk bf16 in WMMA-B operand order
//   k_gemm_h   : h = x @ W^T via v_wmma_f32_16x16x32_bf16, stored transposed
//                hT[b][feat][token] bf16 (feat = h*64+d)        [B,256,2048]
//   k_scores   : s_src/s_dst[b][h][n] fp32
//   k_attn     : flash-style online softmax over j-tiles of 32 with bitmask
//                adjacency; rescale only when the running max grows
//                (wave-uniform branch); P@V accumulated with WMMA (4 d-tiles)
// ---------------------------------------------------------------------------

#define Bq   4
#define Nq   2048
#define Dq   256      // DIN == DOUT
#define Hq   4
#define HDq  64
#define SLOPE 0.2f

typedef __attribute__((ext_vector_type(16))) __bf16 v16bf;
typedef __attribute__((ext_vector_type(8)))  float  v8f;

struct U32x8 { unsigned int d[8]; };

static __device__ __forceinline__ unsigned short f2bf(float f) {
  unsigned int u = __builtin_bit_cast(unsigned int, f);
  unsigned int r = u + 0x7FFFu + ((u >> 16) & 1u);   // round-to-nearest-even
  return (unsigned short)(r >> 16);
}
static __device__ __forceinline__ float bf2f(unsigned short u) {
  unsigned int x = ((unsigned int)u) << 16;
  return __builtin_bit_cast(float, x);
}
// build a v16bf operand from two 16-byte chunks (per-lane contiguous halves)
static __device__ __forceinline__ v16bf load2x8(const unsigned short* p0,
                                                const unsigned short* p1) {
  uint4 a = *(const uint4*)p0;
  uint4 b = *(const uint4*)p1;
  U32x8 u;
  u.d[0] = a.x; u.d[1] = a.y; u.d[2] = a.z; u.d[3] = a.w;
  u.d[4] = b.x; u.d[5] = b.y; u.d[6] = b.z; u.d[7] = b.w;
  return __builtin_bit_cast(v16bf, u);
}
static __device__ __forceinline__ v16bf pack16(const float* p) {
  U32x8 u;
#pragma unroll
  for (int i = 0; i < 8; ++i)
    u.d[i] = (unsigned int)f2bf(p[2 * i]) | ((unsigned int)f2bf(p[2 * i + 1]) << 16);
  return __builtin_bit_cast(v16bf, u);
}

// --------------------------- x fp32 -> bf16 --------------------------------
__global__ void k_cvt_x(const float* __restrict__ x, unsigned short* __restrict__ xb) {
  int i = blockIdx.x * 256 + threadIdx.x;            // 524288 threads, 4 elems each
  float4 v = ((const float4*)x)[i];
  uint2 o;
  o.x = (unsigned int)f2bf(v.x) | ((unsigned int)f2bf(v.y) << 16);
  o.y = (unsigned int)f2bf(v.z) | ((unsigned int)f2bf(v.w) << 16);
  ((uint2*)xb)[i] = o;
}

// ---------------- adj fp32 -> bitmask (one read of adj, coalesced) ---------
__global__ void k_pack_adj(const float* __restrict__ adj,
                           unsigned int* __restrict__ amask) {
  size_t idx = (size_t)blockIdx.x * 256 + threadIdx.x;   // 16.77M threads
  float v = adj[idx];
  unsigned long long bal = __ballot(v != 0.f);           // wave32: bits [31:0]
  if ((threadIdx.x & 31) == 0) amask[idx >> 5] = (unsigned int)bal;
}

// ---------------- W -> WMMA B-operand packed order -------------------------
// wpk[((kt*16 + nt)*32 + lane)*16 + q] = bf16( W[n][k] ),
//   n = nt*16 + (lane&15),  k = kt*32 + ((lane>=16)?8:0) + (q&7) + ((q>=8)?16:0)
__global__ void k_pack_w(const float* __restrict__ W, unsigned short* __restrict__ wpk) {
  int o = blockIdx.x * 256 + threadIdx.x;            // 65536 total
  int q    = o & 15;
  int lane = (o >> 4) & 31;
  int nt   = (o >> 9) & 15;
  int kt   = o >> 13;
  int K = ((lane >> 4) << 3) + (q & 7) + ((q & 8) << 1);
  int n = nt * 16 + (lane & 15);
  int k = kt * 32 + K;
  wpk[o] = f2bf(W[n * Dq + k]);
}

// ---------------- h = x @ W^T, stored transposed as hT[b][feat][token] -----
__global__ void k_gemm_h(const unsigned short* __restrict__ xb,
                         const unsigned short* __restrict__ wpk,
                         unsigned short* __restrict__ hT) {
  int wave = threadIdx.x >> 5, lane = threadIdx.x & 31;
  int tile = blockIdx.x * 8 + wave;                  // 8192 wave tiles
  int mT = tile >> 4, nT = tile & 15;                // 512 x 16 tiles of 16x16
  int r15 = lane & 15, hi = lane >> 4;
  int off8 = hi * 8;

  v8f c;
#pragma unroll
  for (int i = 0; i < 8; ++i) c[i] = 0.f;

  const unsigned short* arow = xb + (size_t)(mT * 16 + r15) * Dq;
#pragma unroll
  for (int kt = 0; kt < 8; ++kt) {
    const unsigned short* ap = arow + kt * 32 + off8;
    const unsigned short* bp = wpk + ((size_t)(kt * 16 + nT) * 32 + lane) * 16;
    v16bf A = load2x8(ap, ap + 16);
    v16bf Bm = load2x8(bp, bp + 8);
    c = __builtin_amdgcn_wmma_f32_16x16x32_bf16(false, A, false, Bm,
                                                (short)0, c, false, false);
  }
  // store: rows m = mT*16 + hi*8 + rr (consecutive tokens), col n fixed
  int b  = (mT * 16) >> 11;
  int t0 = ((mT * 16) & (Nq - 1)) + off8;
  int n  = nT * 16 + r15;
  uint4 st;
  st.x = (unsigned int)f2bf(c[0]) | ((unsigned int)f2bf(c[1]) << 16);
  st.y = (unsigned int)f2bf(c[2]) | ((unsigned int)f2bf(c[3]) << 16);
  st.z = (unsigned int)f2bf(c[4]) | ((unsigned int)f2bf(c[5]) << 16);
  st.w = (unsigned int)f2bf(c[6]) | ((unsigned int)f2bf(c[7]) << 16);
  *(uint4*)(hT + ((size_t)(b * Dq + n)) * Nq + t0) = st;
}

// ---------------- s_src / s_dst : [b][h][n] --------------------------------
__global__ void k_scores(const unsigned short* __restrict__ hT,
                         const float* __restrict__ a_src,
                         const float* __restrict__ a_dst,
                         float* __restrict__ sSrc, float* __restrict__ sDst) {
  int idx = blockIdx.x * 256 + threadIdx.x;          // 32768 = B*H*N, n fastest
  int n  = idx & (Nq - 1);
  int bh = idx >> 11;
  int h  = bh & (Hq - 1);
  const unsigned short* col = hT + (size_t)(bh * HDq) * Nq + n;
  float ss = 0.f, sd = 0.f;
#pragma unroll 8
  for (int d = 0; d < HDq; ++d) {
    float v = bf2f(col[(size_t)d * Nq]);
    ss += v * a_src[h * HDq + d];
    sd += v * a_dst[h * HDq + d];
  }
  sSrc[idx] = ss;
  sDst[idx] = sd;
}

// ---------------- fused masked-softmax + aggregation -----------------------
__global__ void k_attn(const unsigned short* __restrict__ hT,
                       const unsigned int* __restrict__ amask,
                       const float* __restrict__ sSrc,
                       const float* __restrict__ sDst,
                       float* __restrict__ out) {
  int wave = threadIdx.x >> 5, lane = threadIdx.x & 31;
  int head = wave & 3, rowSub = wave >> 2;           // 4 heads x 2 row tiles
  int b  = blockIdx.x >> 6;                          // 64 i-tiles of 32 rows
  int i0 = ((blockIdx.x & 63) << 5) + rowSub * 16;
  int r15 = lane & 15, hi = lane >> 4;
  int off8 = hi * 8, hi16 = hi * 16;
  int bh = b * Hq + head;

  float ssrc_i = sSrc[(size_t)bh * Nq + i0 + r15];   // score of my A-row
  const unsigned int* mrow = amask + ((size_t)(b * Nq + i0 + r15)) * (Nq / 32);
  const float* sdRow = sDst + (size_t)bh * Nq;

  float m_prev = -3.0e38f, lsum = 0.f;
  v8f acc[4];
#pragma unroll
  for (int dt = 0; dt < 4; ++dt)
#pragma unroll
    for (int i = 0; i < 8; ++i) acc[dt][i] = 0.f;

  for (int j0 = 0; j0 < Nq; j0 += 32) {
    unsigned int msk = mrow[j0 >> 5];                // 32 adjacency bits
    float dv[16];
    {
      const float4* q0 = (const float4*)(sdRow + j0 + off8);
      const float4* q1 = (const float4*)(sdRow + j0 + 16 + off8);
      float4 t0 = q0[0], t1 = q0[1], t2 = q1[0], t3 = q1[1];
      dv[0]=t0.x; dv[1]=t0.y; dv[2]=t0.z; dv[3]=t0.w;
      dv[4]=t1.x; dv[5]=t1.y; dv[6]=t1.z; dv[7]=t1.w;
      dv[8]=t2.x; dv[9]=t2.y; dv[10]=t2.z; dv[11]=t2.w;
      dv[12]=t3.x; dv[13]=t3.y; dv[14]=t3.z; dv[15]=t3.w;
    }
    // masked leaky-relu scores, row-local max
    float e[16];
    float mloc = -3.0e38f;
#pragma unroll
    for (int k = 0; k < 16; ++k) {
      int bit = off8 + ((k & 8) << 1) + (k & 7);     // k<8: off8+k, k>=8: off8+16+k-8
      float s = ssrc_i + dv[k];
      s = (s > 0.f) ? s : SLOPE * s;
      e[k] = ((msk >> bit) & 1u) ? s : -3.0e38f;
      mloc = fmaxf(mloc, e[k]);
    }
    float mtile = fmaxf(mloc, __shfl_xor(mloc, 16, 32)); // partner lane holds
    float m_new = fmaxf(m_prev, mtile);                  // the other 16 cols

    // wave-uniform: only rescale when some row's max actually grew (rare)
    if (__any(m_new > m_prev)) {
      float corr = __expf(m_prev - m_new);
      lsum *= corr;
#pragma unroll
      for (int rr = 0; rr < 8; ++rr) {
        float cf = __shfl(corr, hi * 8 + rr, 32);    // C-row m lives on lane m
#pragma unroll
        for (int dt = 0; dt < 4; ++dt) acc[dt][rr] *= cf;
      }
      m_prev = m_new;
    }

    float pk[16], ps = 0.f;
#pragma unroll
    for (int k = 0; k < 16; ++k) { pk[k] = __expf(e[k] - m_prev); ps += pk[k]; }
    ps += __shfl_xor(ps, 16, 32);
    lsum += ps;

    v16bf Aop = pack16(pk);                              // slot order == A order
#pragma unroll
    for (int dt = 0; dt < 4; ++dt) {
      const unsigned short* bp =
          hT + ((size_t)(bh * HDq + dt * 16 + r15)) * Nq + j0 + off8;
      v16bf Bop = load2x8(bp, bp + 16);                  // K = off8+0..7, off8+16..23
      acc[dt] = __builtin_amdgcn_wmma_f32_16x16x32_bf16(false, Aop, false, Bop,
                                                        (short)0, acc[dt],
                                                        false, false);
    }
    if (j0 + 32 < Nq) {
      __builtin_prefetch(sdRow + j0 + 32 + off8, 0, 1);  // global_prefetch_b8
      __builtin_prefetch(mrow + ((j0 + 32) >> 5), 0, 1);
    }
  }

  float linv[8];
#pragma unroll
  for (int rr = 0; rr < 8; ++rr)
    linv[rr] = 1.0f / __shfl(lsum, hi * 8 + rr, 32);

#pragma unroll
  for (int dt = 0; dt < 4; ++dt)
#pragma unroll
    for (int rr = 0; rr < 8; ++rr) {
      int m = hi * 8 + rr;
      out[((size_t)(b * Nq + i0 + m)) * Dq + head * HDq + dt * 16 + r15] =
          acc[dt][rr] * linv[rr];
    }
}

// ---------------------------------------------------------------------------
extern "C" void kernel_launch(void* const* d_in, const int* in_sizes, int n_in,
                              void* d_out, int out_size, void* d_ws, size_t ws_size,
                              hipStream_t stream) {
  (void)in_sizes; (void)n_in; (void)out_size; (void)ws_size;
  const float* x     = (const float*)d_in[0];   // [B,N,DIN]
  const float* adj   = (const float*)d_in[1];   // [B,N,N]
  const float* W     = (const float*)d_in[2];   // [DOUT,DIN]
  const float* a_src = (const float*)d_in[3];   // [H,HD]
  const float* a_dst = (const float*)d_in[4];   // [H,HD]
  float* out = (float*)d_out;                   // [B,N,DOUT] fp32

  char* ws = (char*)d_ws;
  unsigned short* hT  = (unsigned short*)(ws);                         // 4 MiB
  unsigned short* xb  = (unsigned short*)(ws + (4u << 20));            // 4 MiB
  unsigned short* wpk = (unsigned short*)(ws + (8u << 20));            // 128 KiB
  float* sSrc = (float*)(ws + (8u << 20) + (128u << 10));              // 128 KiB
  float* sDst = (float*)(ws + (8u << 20) + (256u << 10));              // 128 KiB
  unsigned int* amask = (unsigned int*)(ws + (8u << 20) + (384u << 10)); // 2 MiB

  k_cvt_x   <<<2048,  256, 0, stream>>>(x, xb);
  k_pack_adj<<<65536, 256, 0, stream>>>(adj, amask);
  k_pack_w  <<<256,   256, 0, stream>>>(W, wpk);
  k_gemm_h  <<<1024,  256, 0, stream>>>(xb, wpk, hT);
  k_scores  <<<128,   256, 0, stream>>>(hT, a_src, a_dst, sSrc, sDst);
  k_attn    <<<256,   256, 0, stream>>>(hT, amask, sSrc, sDst, out);
}
